// MLSTM_66726611911228
// MI455X (gfx1250) — matrix operand order repaired and checked by reading, not compiled, not run on hardware
//
#include <hip/hip_runtime.h>

// Problem dims (fixed by reference)
#define TT 1024
#define BB 128
#define II 128
#define HH 256
#define OO 128
#define KK 32

// Decomposition
#define NWG 16          // workgroups; each owns HS hidden columns
#define HS 16           // hidden columns per WG
#define THREADS 256     // 8 wave32 waves
#define KT 12           // k-tiles of 32 over K = I + H = 384

typedef __attribute__((ext_vector_type(16))) _Float16 v16h;
typedef __attribute__((ext_vector_type(8)))  float    v8f;

// LDS budget (per WG, <= 320KB):
//   hist ring buffer : KK*BB*HS fp32   = 262144 B
//   packed gate W    : KT*4*32*16 f16  =  49152 B
//   packed Wo slice  : 8*32*16 f16     =   8192 B
#define HIST_FLOATS (KK * BB * HS)
#define WG_HALVES   (KT * 4 * 32 * 16)
#define WO_HALVES   (8 * 32 * 16)
#define SMEM_BYTES  (HIST_FLOATS * 4 + (WG_HALVES + WO_HALVES) * 2)

__device__ __forceinline__ float sigm(float x)   { return 1.0f / (1.0f + __expf(-x)); }
__device__ __forceinline__ float tanh_s(float x) { float e = __expf(2.0f * x); return 1.0f - 2.0f / (e + 1.0f); }

// A-operand (16 x 32, f16) fragment in contiguous k order (B packed to match):
// lane<16: rows 0..15, K = kb0+0..15 ; lane>=16: K = kb0+16..31
__device__ __forceinline__ v16h load_a_f16(const _Float16* __restrict__ src, int ld, int mb, int kb0, int lane) {
  int m  = mb + (lane & 15);
  int kb = kb0 + ((lane & 16) ? 16 : 0);
  const _Float16* p = src + (size_t)m * ld + kb;
  v16h a;
#pragma unroll
  for (int j = 0; j < 8; ++j) a[j] = p[j];
#pragma unroll
  for (int j = 0; j < 8; ++j) a[8 + j] = p[8 + j];
  return a;
}

// Device-wide barrier across the NWG persistent workgroups (monotonic counter).
__device__ __forceinline__ void grid_barrier(unsigned* cnt, unsigned target) {
  __threadfence();                    // release: publish this WG's stores at device scope
  __syncthreads();
  if (threadIdx.x == 0) {
    atomicAdd(cnt, 1u);
    while (__hip_atomic_load(cnt, __ATOMIC_RELAXED, __HIP_MEMORY_SCOPE_AGENT) < target)
      __builtin_amdgcn_s_sleep(1);
  }
  __syncthreads();
  __threadfence();                    // acquire side for all threads
}

__global__ void mlstm_init_ws(const float* __restrict__ inputs, unsigned* __restrict__ cnt,
                              _Float16* __restrict__ hbase, _Float16* __restrict__ xfbase) {
  int i = blockIdx.x * blockDim.x + threadIdx.x;
  int stride = gridDim.x * blockDim.x;
  if (i == 0) *cnt = 0u;
  for (int j = i; j < 2 * BB * HH; j += stride) hbase[j] = (_Float16)0.0f;
  // Seed the x-f16 pipeline: x_0 -> buf0, x_1 -> buf1
  for (int j = i; j < BB * II; j += stride) {
    xfbase[j]           = (_Float16)inputs[j];
    xfbase[BB * II + j] = (_Float16)inputs[(size_t)BB * II + j];
  }
}

__global__ __launch_bounds__(THREADS)
void mlstm_persistent(const float* __restrict__ inputs,
                      const float* __restrict__ Wx,  const float* __restrict__ Wh,
                      const float* __restrict__ bvec,
                      const float* __restrict__ Wdx, const float* __restrict__ Wdh,
                      const float* __restrict__ bdv,
                      const float* __restrict__ Wo,  const float* __restrict__ bov,
                      float* __restrict__ out,
                      _Float16* __restrict__ hbase, _Float16* __restrict__ xfbase,
                      unsigned* __restrict__ cnt) {
  extern __shared__ char smem_raw[];
  float*    hist = (float*)smem_raw;                          // [KK][BB][HS] ring buffer
  _Float16* wgW  = (_Float16*)(smem_raw + HIST_FLOATS * 4);   // packed gate B-fragments
  _Float16* woW  = wgW + WG_HALVES;                           // packed Wo B-fragments

  const int gid  = blockIdx.x;        // 0..15 : owns hidden cols [gid*HS, gid*HS+HS)
  const int tid  = threadIdx.x;
  const int lane = tid & 31;
  const int wave = tid >> 5;          // 0..7 : M-tile (16 batch rows)
  const int col  = lane & 15;         // column within this WG's 16-wide slice
  const int hi8  = lane >> 4;         // C-fragment row-half selector
  const int mb   = wave * 16;

  float* y_out   = out;
  float* hf_out  = out + (size_t)TT * BB * OO;
  float* hcf_out = hf_out + (size_t)BB * HH;
  float* df_out  = hcf_out + (size_t)KK * BB * HH;

  // Per-lane loop-invariant biases (registers, not LDS).
  const int hc0 = gid * HS + col;
  const float b_i = bvec[hc0];
  const float b_o = bvec[HH + hc0];
  const float b_g = bvec[2 * HH + hc0];
  const float b_d = bdv[hc0];
  const float b_y = (gid < 8) ? bov[gid * 16 + col] : 0.0f;

  // ---- Init: pre-swizzle weights into WMMA B-operand lane order in LDS ----
  // B-frag layout: halves at ((ktile*4 + n)*32 + lane)*16 + j ;
  // lane<16: N=lane, K=kt*32+0..15 ; lane>=16: N=lane-16, K=kt*32+16..31
  for (int idx = tid; idx < WG_HALVES; idx += THREADS) {
    int j = idx & 15;
    int l = (idx >> 4) & 31;
    int n = (idx >> 9) & 3;           // 0=i, 1=o, 2=g, 3=d-path
    int k = idx >> 11;                // k-tile 0..11
    int c  = l & 15;
    int kk = k * 32 + ((l < 16) ? 0 : 16) + j;
    int hc = gid * HS + c;
    float wv;
    if (kk < II) wv = (n < 3) ? Wx[(size_t)kk * (3 * HH) + n * HH + hc] : Wdx[(size_t)kk * HH + hc];
    else { int hr = kk - II;
           wv = (n < 3) ? Wh[(size_t)hr * (3 * HH) + n * HH + hc] : Wdh[(size_t)hr * HH + hc]; }
    wgW[idx] = (_Float16)wv;
  }
  if (gid < 8) {
    for (int idx = tid; idx < WO_HALVES; idx += THREADS) {
      int j = idx & 15;
      int l = (idx >> 4) & 31;
      int k = idx >> 9;               // k-tile 0..7 over H=256
      int c  = l & 15;
      int kk = k * 32 + ((l < 16) ? 0 : 16) + j;
      woW[idx] = (_Float16)Wo[(size_t)kk * OO + gid * 16 + c];
    }
  }
  for (int idx = tid; idx < HIST_FLOATS; idx += THREADS) hist[idx] = 0.0f;
  __syncthreads();

  // ---- Sequential time loop (1 device barrier per step) ----
  for (int t = 0; t < TT; ++t) {
    // Single global base + integer offset (keeps loads in the global, not flat, path).
    const size_t hOffIn  = (size_t)(t & 1) * (BB * HH);
    const size_t hOffOut = (size_t)((t + 1) & 1) * (BB * HH);
    const size_t xOff    = (size_t)(t & 1) * (BB * II);

    // Phase A: gates/d GEMM.  C tile grid: wave => M-tile, n => {i,o,g,d} 16-col slice.
    v8f acc[4] = {};
#pragma unroll
    for (int k = 0; k < KT; ++k) {
      v16h a = (k < 4) ? load_a_f16(xfbase + xOff, II, mb, k * 32, lane)
                       : load_a_f16(hbase + hOffIn, HH, mb, (k - 4) * 32, lane);
#pragma unroll
      for (int n = 0; n < 4; ++n) {
        v16h bfr = *(const v16h*)(wgW + ((size_t)(k * 4 + n) * 32 + lane) * 16);
        acc[n] = __builtin_amdgcn_wmma_f32_16x16x32_f16(false, a, false, bfr,
                                                        (short)0, acc[n], false, false);
      }
    }

    // Activations + fractional-memory filter (history is LDS-resident ring buffer).
    const int slotW = t & (KK - 1);
#pragma unroll
    for (int r = 0; r < 8; ++r) {
      int bb = mb + r + hi8 * 8;      // batch row this lane owns for C-fragment row r
      float ig = sigm(acc[0][r] + b_i);
      float og = sigm(acc[1][r] + b_o);
      float gg = tanh_s(acc[2][r] + b_g);
      float dv = 0.5f * sigm(acc[3][r] + b_d);

      const float* hrow = hist + bb * HS + col;
      float c = ig * gg;
      float w = dv;                   // w_1 = d
#pragma unroll
      for (int ag = 0; ag < KK; ++ag) {
        int slot = (t - 1 - ag) & (KK - 1);                 // age ag -> ring slot
        c += w * hrow[slot * (BB * HS)];
        // w_{j+1} = w_j * (j - d)/(j+1), j = ag+1 : one fma + one mul
        const float C1 = (float)(ag + 1) / (float)(ag + 2);
        const float C2 = 1.0f / (float)(ag + 2);
        w *= (C1 - dv * C2);
      }
      float hn = og * tanh_s(c);

      hist[slotW * (BB * HS) + bb * HS + col] = c;          // newest cell state
      hbase[hOffOut + (size_t)bb * HH + gid * HS + col] = (_Float16)hn;
      if (t == TT - 1) {
        hf_out[(size_t)bb * HH + gid * HS + col] = hn;
        df_out[(size_t)bb * HH + gid * HS + col] = dv;
      }
    }

    // All WGs must publish their h slice before anyone uses full h.
    grid_barrier(cnt, (unsigned)(NWG * (t + 1)));

    if (gid < 8) {
      // Phase B: y_t = h_new @ Wo + bo.  WGs 0..7 each own 16 output columns.
      v8f yacc = {};
#pragma unroll
      for (int k = 0; k < 8; ++k) {
        v16h a2  = load_a_f16(hbase + hOffOut, HH, mb, k * 32, lane);
        v16h bf2 = *(const v16h*)(woW + ((size_t)k * 32 + lane) * 16);
        yacc = __builtin_amdgcn_wmma_f32_16x16x32_f16(false, a2, false, bf2,
                                                      (short)0, yacc, false, false);
      }
      float* yrow = y_out + (size_t)t * BB * OO;
#pragma unroll
      for (int r = 0; r < 8; ++r) {
        int bb = mb + r + hi8 * 8;
        yrow[(size_t)bb * OO + gid * 16 + col] = yacc[r] + b_y;
      }
    } else {
      // WGs 8..15: pipeline x_{t+2} f32 -> f16 into buffer (t+2)&1 == t&1.
      // Safe: every WG finished reading xfbase[t&1 buf] (phase A of step t) before
      // barrier t completed; we only write after passing barrier t.
      if (t + 2 < TT) {
        const float* xsrc = inputs + (size_t)(t + 2) * BB * II;
        _Float16*    xdst = xfbase + xOff;             // (t+2)&1 == t&1
        const int chunk = (BB * II) / 8;               // 2048 elems per WG
        const int base  = (gid - 8) * chunk;
        if (t + 3 < TT)
          __builtin_prefetch(inputs + (size_t)(t + 3) * BB * II + base + tid * 8, 0, 1);
        for (int i2 = tid; i2 < chunk; i2 += THREADS)
          xdst[base + i2] = (_Float16)xsrc[base + i2];
      }
    }
  }

  // ---- Final history writeout: ring -> newest-first order ----
  for (int idx = tid; idx < KK * BB * HS; idx += THREADS) {
    int c2 = idx % HS;
    int b2 = (idx / HS) % BB;
    int a2 = idx / (HS * BB);                               // age
    int slot = (TT - 1 - a2) & (KK - 1);
    hcf_out[(size_t)a2 * BB * HH + (size_t)b2 * HH + gid * HS + c2] =
        hist[slot * (BB * HS) + b2 * HS + c2];
  }
}

extern "C" void kernel_launch(void* const* d_in, const int* in_sizes, int n_in,
                              void* d_out, int out_size, void* d_ws, size_t ws_size,
                              hipStream_t stream) {
  (void)in_sizes; (void)n_in; (void)out_size; (void)ws_size;
  const float* inputs = (const float*)d_in[0];
  const float* Wx     = (const float*)d_in[1];
  const float* Wh     = (const float*)d_in[2];
  const float* bvec   = (const float*)d_in[3];
  const float* Wdx    = (const float*)d_in[4];
  const float* Wdh    = (const float*)d_in[5];
  const float* bdv    = (const float*)d_in[6];
  const float* Wo     = (const float*)d_in[7];
  const float* bov    = (const float*)d_in[8];

  unsigned*  cnt    = (unsigned*)d_ws;
  _Float16*  hbase  = (_Float16*)((char*)d_ws + 256);                       // 2 x (B x H) f16
  _Float16*  xfbase = (_Float16*)((char*)d_ws + 256 + 2 * BB * HH * 2);     // 2 x (B x I) f16

  mlstm_init_ws<<<128, 256, 0, stream>>>(inputs, cnt, hbase, xfbase);
  mlstm_persistent<<<NWG, THREADS, SMEM_BYTES, stream>>>(
      inputs, Wx, Wh, bvec, Wdx, Wdh, bdv, Wo, bov,
      (float*)d_out, hbase, xfbase, cnt);
}
